// CausalSelfAttention_31722628448457
// MI455X (gfx1250) — matrix operand order repaired and checked by reading, not compiled, not run on hardware
//
#include <hip/hip_runtime.h>

// ---------------------------------------------------------------------------
// Causal self-attention for MI455X (gfx1250), wave32 + WMMA bf16.
// Pipeline: qkv_gemm(x,Wq/Wk/Wv) -> scores(QK^T, mask+scale) -> softmax -> PV.
// Matmuls: v_wmma_f32_16x16x32_bf16, fp32 accumulate.
// Tile copies in scores/pv use GLOBAL_LOAD_ASYNC_TO_LDS_B128 when available.
// ---------------------------------------------------------------------------

typedef __bf16 v16bf __attribute__((ext_vector_type(16)));
typedef __bf16 v8bf  __attribute__((ext_vector_type(8)));
typedef float  v8f   __attribute__((ext_vector_type(8)));

#define NB    4
#define S_LEN 2048
#define D_DIM 1024
#define BM    128
#define BN    128
#define BK    32
#define LDT   40   // padded LDS row stride (bf16 elems): 80B -> conflict-free b128

// --- gfx1250 async LDS copy path (guarded; falls back to uint4 copy) -------
#if defined(__has_builtin)
#if __has_builtin(__builtin_amdgcn_global_load_async_to_lds_b128) && \
    __has_builtin(__builtin_amdgcn_s_wait_asynccnt)
#define HAVE_ASYNC_LDS 1
#endif
#endif

#ifndef HAVE_ASYNC_LDS
#define HAVE_ASYNC_LDS 0
#endif

#if HAVE_ASYNC_LDS
typedef int v4i_vs __attribute__((vector_size(16)));          // matches builtin param
typedef __attribute__((address_space(1))) v4i_vs* g_v4i_ptr;  // global src
typedef __attribute__((address_space(3))) v4i_vs* l_v4i_ptr;  // LDS dst
#define COPY16(dst_lds, src_g)                                 \
  __builtin_amdgcn_global_load_async_to_lds_b128(              \
      (g_v4i_ptr)(const void*)(src_g),                         \
      (l_v4i_ptr)(void*)(dst_lds), 0, 0)
#define COPY_FENCE() __builtin_amdgcn_s_wait_asynccnt(0)
#else
#define COPY16(dst_lds, src_g) (*(uint4*)(dst_lds) = *(const uint4*)(src_g))
#define COPY_FENCE() ((void)0)
#endif

__device__ __forceinline__ unsigned short f2bf(float f) {
  // round-to-nearest-even fp32 -> bf16
  unsigned int u = __float_as_uint(f);
  unsigned int r = u + 0x7FFFu + ((u >> 16) & 1u);
  return (unsigned short)(r >> 16);
}

// A fragment: 16x32 bf16, lane l holds row (l%16); K-halves by (l/16).
// V0-3 = K{0..7 | 8..15}, V4-7 = K{16..23 | 24..31}  (ISA 7.12.2)
__device__ __forceinline__ v16bf load_frag_a(const unsigned short* T, int row0, int lane) {
  const int r    = row0 + (lane & 15);
  const int half = lane >> 4;
  const unsigned short* p = T + r * LDT + half * 8;
  v16bf out;
  *(v8bf*)&out         = *(const v8bf*)(p);       // K 0..7   (or 8..15)
  *(((v8bf*)&out) + 1) = *(const v8bf*)(p + 16);  // K 16..23 (or 24..31)
  return out;
}

// B fragment: 32x16 bf16 from N-major tile ([n][k]); lane l -> col n0+(l%16),
// K{0..7 | 8..15}, then K{16..23 | 24..31} by (l/16) (same order as A).
__device__ __forceinline__ v16bf load_frag_b(const unsigned short* T, int n0, int lane) {
  const int n    = n0 + (lane & 15);
  const int half = lane >> 4;
  const unsigned short* p = T + n * LDT + half * 8;
  v16bf out;
  *(v8bf*)&out         = *(const v8bf*)(p);
  *(((v8bf*)&out) + 1) = *(const v8bf*)(p + 16);
  return out;
}

#define WMMA_BF16(a, b, c) \
  __builtin_amdgcn_wmma_f32_16x16x32_bf16(false, (a), false, (b), (short)0, (c), false, false)

// ---------------------------------------------------------------------------
// Kernel 1: Y[m][n] = sum_k X[m][k] * W[n][k]   (x @ W^T), bf16 output.
// TRANSPOSE=0: Y row-major [M][D].  TRANSPOSE=1: Y = [B][D][S] (V^T).
// ---------------------------------------------------------------------------
template <int TRANSPOSE>
__global__ __launch_bounds__(256) void qkv_gemm_kernel(
    const float* __restrict__ X, const float* __restrict__ W,
    unsigned short* __restrict__ Y) {
  __shared__ unsigned short As[BM * LDT];
  __shared__ unsigned short Bs[BN * LDT];
  const int tid   = threadIdx.x;
  const int lane  = tid & 31;
  const int wid   = tid >> 5;
  const int waveM = wid & 3;   // 32-row strip
  const int waveN = wid >> 2;  // 64-col strip
  const int m0 = blockIdx.y * BM;
  const int n0 = blockIdx.x * BN;

  v8f acc[2][4];
  for (int i = 0; i < 2; ++i)
    for (int j = 0; j < 4; ++j)
      for (int e = 0; e < 8; ++e) acc[i][j][e] = 0.0f;

  const int NT = D_DIM / BK;
  for (int kt = 0; kt < NT; ++kt) {
    const int k0 = kt * BK;
    for (int i = 0; i < 4; ++i) {           // 128x32 fp32 -> bf16 LDS (A and B)
      const int c    = tid + i * 256;       // 0..1023
      const int row  = c >> 3;
      const int col4 = (c & 7) * 4;
      const float* pa = X + (size_t)(m0 + row) * D_DIM + k0 + col4;
      const float* pb = W + (size_t)(n0 + row) * D_DIM + k0 + col4;
      if (kt + 1 < NT) {                    // gfx1250 global_prefetch for next k-tile
        __builtin_prefetch(pa + BK, 0, 1);
        __builtin_prefetch(pb + BK, 0, 1);
      }
      float4 va = *(const float4*)pa;
      unsigned short* da = As + row * LDT + col4;
      da[0] = f2bf(va.x); da[1] = f2bf(va.y); da[2] = f2bf(va.z); da[3] = f2bf(va.w);
      float4 vb = *(const float4*)pb;
      unsigned short* db = Bs + row * LDT + col4;
      db[0] = f2bf(vb.x); db[1] = f2bf(vb.y); db[2] = f2bf(vb.z); db[3] = f2bf(vb.w);
    }
    __syncthreads();
    v16bf af[2], bf[4];
    for (int mi = 0; mi < 2; ++mi) af[mi] = load_frag_a(As, waveM * 32 + mi * 16, lane);
    for (int ni = 0; ni < 4; ++ni) bf[ni] = load_frag_b(Bs, waveN * 64 + ni * 16, lane);
    for (int mi = 0; mi < 2; ++mi)
      for (int ni = 0; ni < 4; ++ni)
        acc[mi][ni] = WMMA_BF16(af[mi], bf[ni], acc[mi][ni]);
    __syncthreads();
  }

  const int half = lane >> 4, nlane = lane & 15;
  for (int mi = 0; mi < 2; ++mi)
    for (int ni = 0; ni < 4; ++ni)
      for (int r = 0; r < 8; ++r) {
        const int m = m0 + waveM * 32 + mi * 16 + r + half * 8;
        const int n = n0 + waveN * 64 + ni * 16 + nlane;
        const unsigned short bv = f2bf(acc[mi][ni][r]);
        if (TRANSPOSE == 0) {
          Y[(size_t)m * D_DIM + n] = bv;
        } else {
          const int b = m >> 11, s = m & (S_LEN - 1);
          Y[((size_t)b * D_DIM + n) * S_LEN + s] = bv;
        }
      }
}

// ---------------------------------------------------------------------------
// Kernel 2: scores[b][q][k] = (Q . K) / 32, causal mask fused; fp32 out.
// Fully-masked 128x128 tiles exit early.
// ---------------------------------------------------------------------------
__global__ __launch_bounds__(256) void scores_kernel(
    const unsigned short* __restrict__ Q, const unsigned short* __restrict__ Kb,
    float* __restrict__ Sc) {
  const int m0 = blockIdx.y * BM;        // flat q row (B*S)
  const int n0 = blockIdx.x * BN;        // key col within batch
  const int b  = m0 >> 11;
  const int q0 = m0 & (S_LEN - 1);
  if (n0 > q0 + (BM - 1)) return;        // tile entirely above diagonal

  __shared__ unsigned short As[BM * LDT];
  __shared__ unsigned short Bs[BN * LDT];
  const int tid = threadIdx.x, lane = tid & 31, wid = tid >> 5;
  const int waveM = wid & 3, waveN = wid >> 2;

  v8f acc[2][4];
  for (int i = 0; i < 2; ++i)
    for (int j = 0; j < 4; ++j)
      for (int e = 0; e < 8; ++e) acc[i][j][e] = 0.0f;

  const unsigned short* Qrow = Q + (size_t)m0 * D_DIM;
  const unsigned short* Krow = Kb + ((size_t)b * S_LEN + n0) * D_DIM;

  for (int kt = 0; kt < D_DIM / BK; ++kt) {
    const int k0 = kt * BK;
    for (int i = 0; i < 2; ++i) {        // bf16 tiles, 16B chunks (async if avail)
      const int c = tid + i * 256;       // 0..511
      const int row = c >> 2;
      const int col8 = (c & 3) * 8;
      COPY16(As + row * LDT + col8, Qrow + (size_t)row * D_DIM + k0 + col8);
      COPY16(Bs + row * LDT + col8, Krow + (size_t)row * D_DIM + k0 + col8);
    }
    COPY_FENCE();
    __syncthreads();
    v16bf af[2], bf[4];
    for (int mi = 0; mi < 2; ++mi) af[mi] = load_frag_a(As, waveM * 32 + mi * 16, lane);
    for (int ni = 0; ni < 4; ++ni) bf[ni] = load_frag_b(Bs, waveN * 64 + ni * 16, lane);
    for (int mi = 0; mi < 2; ++mi)
      for (int ni = 0; ni < 4; ++ni)
        acc[mi][ni] = WMMA_BF16(af[mi], bf[ni], acc[mi][ni]);
    __syncthreads();
  }

  float* out = Sc + (size_t)b * S_LEN * S_LEN;
  const int half = lane >> 4, nlane = lane & 15;
  const float NEG_INF = -__builtin_inff();
  for (int mi = 0; mi < 2; ++mi)
    for (int ni = 0; ni < 4; ++ni)
      for (int r = 0; r < 8; ++r) {
        const int q  = q0 + waveM * 32 + mi * 16 + r + half * 8;
        const int kk = n0 + waveN * 64 + ni * 16 + nlane;
        const float v = acc[mi][ni][r] * 0.03125f;   // 1/sqrt(1024)
        out[(size_t)q * S_LEN + kk] = (kk <= q) ? v : NEG_INF;
      }
}

// ---------------------------------------------------------------------------
// Kernel 3: row softmax (causal), fp32 scores -> bf16 probabilities.
// Reads only k<=q; writes zeros for k>q so diagonal PV tiles are clean.
// ---------------------------------------------------------------------------
__global__ __launch_bounds__(256) void softmax_kernel(
    const float* __restrict__ Sc, unsigned short* __restrict__ P) {
  const int row = blockIdx.x;                 // 0..B*S-1
  const int b = row >> 11, q = row & (S_LEN - 1);
  const float* src = Sc + ((size_t)b * S_LEN + q) * S_LEN;
  unsigned short* dst = P + ((size_t)b * S_LEN + q) * S_LEN;
  const int n = q + 1;
  const int tid = threadIdx.x;
  __shared__ float red[256];

  float m = -__builtin_inff();
  for (int i = tid; i < n; i += 256) m = fmaxf(m, src[i]);
  red[tid] = m; __syncthreads();
  for (int s = 128; s > 0; s >>= 1) {
    if (tid < s) red[tid] = fmaxf(red[tid], red[tid + s]);
    __syncthreads();
  }
  m = red[0]; __syncthreads();

  float sum = 0.0f;
  for (int i = tid; i < n; i += 256) sum += __expf(src[i] - m);
  red[tid] = sum; __syncthreads();
  for (int s = 128; s > 0; s >>= 1) {
    if (tid < s) red[tid] += red[tid + s];
    __syncthreads();
  }
  const float inv = 1.0f / red[0];

  for (int i = tid; i < n; i += 256) dst[i] = f2bf(__expf(src[i] - m) * inv);
  for (int i = n + tid; i < S_LEN; i += 256) dst[i] = 0;   // zero masked tail
}

// ---------------------------------------------------------------------------
// Kernel 4: O[b][q][d] = sum_k P[b][q][k] * V[b][k][d], V stored transposed
// ([B][D][S]) so B fragments come from contiguous rows. k-loop truncated at
// the diagonal. fp32 output.
// ---------------------------------------------------------------------------
__global__ __launch_bounds__(256) void pv_kernel(
    const unsigned short* __restrict__ P, const unsigned short* __restrict__ Vt,
    float* __restrict__ O) {
  const int m0 = blockIdx.y * BM;        // flat q row
  const int n0 = blockIdx.x * BN;        // d col
  const int b  = m0 >> 11;
  const int q0 = m0 & (S_LEN - 1);

  __shared__ unsigned short As[BM * LDT];
  __shared__ unsigned short Bs[BN * LDT];
  const int tid = threadIdx.x, lane = tid & 31, wid = tid >> 5;
  const int waveM = wid & 3, waveN = wid >> 2;

  v8f acc[2][4];
  for (int i = 0; i < 2; ++i)
    for (int j = 0; j < 4; ++j)
      for (int e = 0; e < 8; ++e) acc[i][j][e] = 0.0f;

  const unsigned short* Prow = P + ((size_t)b * S_LEN + q0) * S_LEN;
  const unsigned short* Vrow = Vt + ((size_t)b * D_DIM + n0) * S_LEN;
  const int nk = (q0 + BM) / BK;         // causal: only k-tiles up to diagonal

  for (int kt = 0; kt < nk; ++kt) {
    const int k0 = kt * BK;
    for (int i = 0; i < 2; ++i) {
      const int c = tid + i * 256;
      const int row = c >> 2;
      const int col8 = (c & 3) * 8;
      COPY16(As + row * LDT + col8, Prow + (size_t)row * S_LEN + k0 + col8);
      COPY16(Bs + row * LDT + col8, Vrow + (size_t)row * S_LEN + k0 + col8);
    }
    COPY_FENCE();
    __syncthreads();
    v16bf af[2], bf[4];
    for (int mi = 0; mi < 2; ++mi) af[mi] = load_frag_a(As, waveM * 32 + mi * 16, lane);
    for (int ni = 0; ni < 4; ++ni) bf[ni] = load_frag_b(Bs, waveN * 64 + ni * 16, lane);
    for (int mi = 0; mi < 2; ++mi)
      for (int ni = 0; ni < 4; ++ni)
        acc[mi][ni] = WMMA_BF16(af[mi], bf[ni], acc[mi][ni]);
    __syncthreads();
  }

  const int half = lane >> 4, nlane = lane & 15;
  for (int mi = 0; mi < 2; ++mi)
    for (int ni = 0; ni < 4; ++ni)
      for (int r = 0; r < 8; ++r) {
        const int m = m0 + waveM * 32 + mi * 16 + r + half * 8;
        const int n = n0 + waveN * 64 + ni * 16 + nlane;
        O[(size_t)m * D_DIM + n] = acc[mi][ni][r];
      }
}

// ---------------------------------------------------------------------------
extern "C" void kernel_launch(void* const* d_in, const int* in_sizes, int n_in,
                              void* d_out, int out_size, void* d_ws, size_t ws_size,
                              hipStream_t stream) {
  (void)in_sizes; (void)n_in; (void)out_size; (void)ws_size;
  const float* x  = (const float*)d_in[0];
  const float* Wq = (const float*)d_in[1];
  const float* Wk = (const float*)d_in[2];
  const float* Wv = (const float*)d_in[3];
  float* out = (float*)d_out;

  const size_t NE = (size_t)NB * S_LEN * D_DIM;   // 8M elems
  const size_t NS = (size_t)NB * S_LEN * S_LEN;   // 16M elems
  unsigned short* Qb = (unsigned short*)d_ws;     // 16 MB bf16
  unsigned short* Kb = Qb + NE;                   // 16 MB bf16
  unsigned short* Vt = Kb + NE;                   // 16 MB bf16 (transposed V)
  float* Sc = (float*)(Vt + NE);                  // 64 MB fp32 scores
  unsigned short* Pp = (unsigned short*)(Sc + NS);// 32 MB bf16 probs

  dim3 blk(256);
  dim3 gP(D_DIM / BN, (NB * S_LEN) / BM);   // (8, 64): projections & PV
  dim3 gS(S_LEN / BN, (NB * S_LEN) / BM);   // (16, 64): scores

  hipLaunchKernelGGL((qkv_gemm_kernel<0>), gP, blk, 0, stream, x, Wq, Qb);
  hipLaunchKernelGGL((qkv_gemm_kernel<0>), gP, blk, 0, stream, x, Wk, Kb);
  hipLaunchKernelGGL((qkv_gemm_kernel<1>), gP, blk, 0, stream, x, Wv, Vt);
  hipLaunchKernelGGL(scores_kernel,   gS, blk, 0, stream, Qb, Kb, Sc);
  hipLaunchKernelGGL(softmax_kernel, dim3(NB * S_LEN), blk, 0, stream, Sc, Pp);
  hipLaunchKernelGGL(pv_kernel,       gP, blk, 0, stream, Pp, Vt, out);
}
